// ResidualLIFBlock_78022375899806
// MI455X (gfx1250) — matrix operand order repaired and checked by reading, not compiled, not run on hardware
//
#include <hip/hip_runtime.h>
#include <hip/hip_bf16.h>
#include <math.h>

typedef _Float16 v16h __attribute__((ext_vector_type(16)));
typedef _Float16 v8h  __attribute__((ext_vector_type(8)));
typedef float    v8f  __attribute__((ext_vector_type(8)));

#define NB   16    // batch
#define CIN  64
#define COUT 128
#define HH   64
#define WW   64
#define DT_C      0.01f
#define MIN_TAU_C 0.1f

#define XSTRIDE  72    // halves per (row,col) slot in x LDS tile (16B aligned, low bank conflict)
#define S1STRIDE 136   // halves per position in s1 LDS tile (272B = 17*16, 16B aligned)

// ---------------- scratch layout (bytes) ----------------
#define OFF_W1P 0                         // 9*128*64 f16 = 147456
#define OFF_W2H 147456                    // 128*128 f16  = 32768
#define OFF_WSH 180224                    // 128*64 f16   = 16384
#define OFF_A1  196608                    // 128 f32
#define OFF_A2  197120
#define OFF_AS  197632

// ---------------- prep: f16 weight repack + alpha ----------------
__global__ void resid_lif_prep(const float* __restrict__ w1, const float* __restrict__ tau1,
                               const float* __restrict__ w2, const float* __restrict__ tau2,
                               const float* __restrict__ ws, const float* __restrict__ taus,
                               _Float16* __restrict__ w1p, _Float16* __restrict__ w2h,
                               _Float16* __restrict__ wsh,
                               float* __restrict__ a1, float* __restrict__ a2,
                               float* __restrict__ as_) {
  const int tid    = blockIdx.x * blockDim.x + threadIdx.x;
  const int stride = gridDim.x * blockDim.x;
  // w1p[t][m][c] = w1[m][c*9 + t]   (tap-major so GEMM K runs over contiguous c)
  for (int idx = tid; idx < 9 * COUT * CIN; idx += stride) {
    int t   = idx / (COUT * CIN);
    int rem = idx - t * (COUT * CIN);
    int m   = rem >> 6;
    int c   = rem & 63;
    w1p[idx] = (_Float16)w1[m * 576 + c * 9 + t];
  }
  for (int idx = tid; idx < COUT * COUT; idx += stride) w2h[idx] = (_Float16)w2[idx];
  for (int idx = tid; idx < COUT * CIN;  idx += stride) wsh[idx] = (_Float16)ws[idx];
  if (tid < COUT) {
    a1[tid]  = DT_C / (MIN_TAU_C + 1.0f / (1.0f + expf(-tau1[tid])));
    a2[tid]  = DT_C / (MIN_TAU_C + 1.0f / (1.0f + expf(-tau2[tid])));
    as_[tid] = DT_C / (MIN_TAU_C + 1.0f / (1.0f + expf(-taus[tid])));
  }
}

// ---------------- fragment helpers (CDNA5 16x16x32 f16 layouts) ----------------
// A (16x32): lane = M row; elements 0..15 -> K = 16*hi + e  (contiguous)
__device__ __forceinline__ v16h fragA(const _Float16* p, int hi) {
  const v8h lo = *reinterpret_cast<const v8h*>(p + 16 * hi);
  const v8h h2 = *reinterpret_cast<const v8h*>(p + 16 * hi + 8);
  v16h r;
#pragma unroll
  for (int e = 0; e < 8; ++e) { r[e] = lo[e]; r[e + 8] = h2[e]; }
  return r;
}
// B (32x16): lane = N col; elements 0..15 -> K = 16*hi + e  (contiguous)
__device__ __forceinline__ v16h fragB(const _Float16* p) {  // caller adds 16*hi
  const v8h lo = *reinterpret_cast<const v8h*>(p);
  const v8h h2 = *reinterpret_cast<const v8h*>(p + 8);
  v16h r;
#pragma unroll
  for (int e = 0; e < 8; ++e) { r[e] = lo[e]; r[e + 8] = h2[e]; }
  return r;
}

__device__ __forceinline__ v8f wmma_f16(v16h a, v16h b, v8f c) {
  return __builtin_amdgcn_wmma_f32_16x16x32_f16(false, a, false, b, (short)0, c, false, false);
}

// ---------------- fused ResidualLIFBlock kernel ----------------
// grid = NB*HH workgroups (one image row each), 256 threads = 8 waves.
// wave w owns output channels [16w, 16w+16); workgroup covers 64 positions = 4 N-tiles.
__global__ void __launch_bounds__(256)
resid_lif_fused(const float* __restrict__ x,
                const _Float16* __restrict__ w1p, const _Float16* __restrict__ w2h,
                const _Float16* __restrict__ wsh,
                const float* __restrict__ a1, const float* __restrict__ a2,
                const float* __restrict__ as_,
                float* __restrict__ out) {
  __shared__ _Float16 smx[3 * 66 * XSTRIDE];   // x halo tile: [r=0..2][col=0..65][c], f16
  __shared__ _Float16 sms1[WW * S1STRIDE];     // s1 spikes:   [pos][c], f16

  const int b  = blockIdx.x >> 6;
  const int r0 = blockIdx.x & 63;

  // ---- stage x halo tile (rows r0-1..r0+1, cols -1..64) into LDS, f32 -> f16, channel-fastest
  for (int idx = threadIdx.x; idx < CIN * 3 * 66; idx += 256) {
    int c   = idx / (3 * 66);
    int rem = idx - c * (3 * 66);
    int r   = rem / 66;
    int col = rem - r * 66;
    int gr  = r0 - 1 + r;
    int gc  = col - 1;
    float v = 0.0f;
    if ((unsigned)gr < (unsigned)HH && (unsigned)gc < (unsigned)WW)
      v = x[((b * CIN + c) * HH + gr) * WW + gc];
    smx[(r * 66 + col) * XSTRIDE + c] = (_Float16)v;
  }
  __syncthreads();

  const int lane  = threadIdx.x & 31;
  const int wave  = threadIdx.x >> 5;
  const int mbase = wave * 16;
  const int hi    = (lane >> 4) & 1;
  const int nloc  = lane & 15;
  const int mrow  = mbase + nloc;   // A-fragment row for this lane
  const int chb   = mbase + 8 * hi; // first output channel this lane's D elements map to

  // per-channel alphas for the 8 channels this lane's accumulator elements touch
  float al1[8], al2[8], als[8];
#pragma unroll
  for (int v = 0; v < 8; ++v) {
    al1[v] = a1[chb + v];
    al2[v] = a2[chb + v];
    als[v] = as_[chb + v];
  }

  float identv[4][8];

  // ---- phase A: conv1 (3x3, K=576 as 9 taps x 2 ksteps) + shortcut, spike, s1 -> LDS
#pragma unroll
  for (int n = 0; n < 4; ++n) {
    const int q = n * 16 + nloc;  // column position in the row
    v8f acc1 = {0.f, 0.f, 0.f, 0.f, 0.f, 0.f, 0.f, 0.f};
#pragma unroll
    for (int t = 0; t < 9; ++t) {
      const int ti = t / 3, tj = t % 3;
      const _Float16* bbase = &smx[(ti * 66 + q + tj) * XSTRIDE];
#pragma unroll
      for (int ks = 0; ks < 2; ++ks) {
        v16h a  = fragA(&w1p[(t * COUT + mrow) * CIN + ks * 32], hi);
        v16h bf = fragB(bbase + ks * 32 + 16 * hi);
        acc1 = wmma_f16(a, bf, acc1);
      }
    }
    v8f acc2 = {0.f, 0.f, 0.f, 0.f, 0.f, 0.f, 0.f, 0.f};
    const _Float16* cb = &smx[(1 * 66 + q + 1) * XSTRIDE];  // center tap
#pragma unroll
    for (int ks = 0; ks < 2; ++ks) {
      v16h a  = fragA(&wsh[mrow * CIN + ks * 32], hi);
      v16h bf = fragB(cb + ks * 32 + 16 * hi);
      acc2 = wmma_f16(a, bf, acc2);
    }
    v8h s1pack;
#pragma unroll
    for (int v = 0; v < 8; ++v) {
      s1pack[v]    = (_Float16)((al1[v] * acc1[v] - 1.0f > 0.0f) ? 1.0f : 0.0f);
      identv[n][v] = (als[v] * acc2[v] - 1.0f > 0.0f) ? 1.0f : 0.0f;
    }
    *reinterpret_cast<v8h*>(&sms1[q * S1STRIDE + chb]) = s1pack;
  }
  __syncthreads();

  // ---- phase B: conv2 (1x1, K=128 = 4 ksteps) on binary spikes, spike, add shortcut, store
#pragma unroll
  for (int n = 0; n < 4; ++n) {
    const int q = n * 16 + nloc;
    v8f acc = {0.f, 0.f, 0.f, 0.f, 0.f, 0.f, 0.f, 0.f};
#pragma unroll
    for (int ks = 0; ks < 4; ++ks) {
      v16h a  = fragA(&w2h[mrow * COUT + ks * 32], hi);
      v16h bf = fragB(&sms1[q * S1STRIDE + ks * 32 + 16 * hi]);
      acc = wmma_f16(a, bf, acc);
    }
#pragma unroll
    for (int v = 0; v < 8; ++v) {
      const int ch = chb + v;
      float o = ((al2[v] * acc[v] - 1.0f > 0.0f) ? 1.0f : 0.0f) + identv[n][v];
      out[((b * COUT + ch) * HH + r0) * WW + q] = o;
    }
  }
}

extern "C" void kernel_launch(void* const* d_in, const int* in_sizes, int n_in,
                              void* d_out, int out_size, void* d_ws, size_t ws_size,
                              hipStream_t stream) {
  const float* x    = (const float*)d_in[0];
  const float* w1   = (const float*)d_in[1];
  const float* tau1 = (const float*)d_in[2];
  const float* w2   = (const float*)d_in[3];
  const float* tau2 = (const float*)d_in[4];
  const float* ws   = (const float*)d_in[5];
  const float* taus = (const float*)d_in[6];
  float* out = (float*)d_out;

  char* wsb = (char*)d_ws;
  _Float16* w1p = (_Float16*)(wsb + OFF_W1P);
  _Float16* w2h = (_Float16*)(wsb + OFF_W2H);
  _Float16* wsh = (_Float16*)(wsb + OFF_WSH);
  float*    a1  = (float*)(wsb + OFF_A1);
  float*    a2  = (float*)(wsb + OFF_A2);
  float*    as_ = (float*)(wsb + OFF_AS);

  resid_lif_prep<<<64, 256, 0, stream>>>(w1, tau1, w2, tau2, ws, taus,
                                         w1p, w2h, wsh, a1, a2, as_);
  resid_lif_fused<<<NB * HH, 256, 0, stream>>>(x, w1p, w2h, wsh, a1, a2, as_, out);
}
